// ConformerBlock_1864015807179
// MI455X (gfx1250) — matrix operand order repaired and checked by reading, not compiled, not run on hardware
//
#include <hip/hip_runtime.h>
#include <math.h>

#define BB 8
#define SS 1024
#define DD 512
#define HH 8
#define DHH 64
#define FFD 2048
#define INNERD 1024
#define KW 31

#define TM 64   // macro-tile M per workgroup
#define TN 64   // macro-tile N per workgroup
#define TK 32   // K step (one WMMA depth)

typedef __attribute__((ext_vector_type(16))) __bf16 v16bf;
typedef __attribute__((ext_vector_type(8)))  __bf16 v8bf;
typedef __attribute__((ext_vector_type(8)))  float  v8f;

// ---------------------------------------------------------------------------
// Workgroup-cooperative strided GEMM:
//   Out[m,n] = act( sum_k (A[m,k] (+ aadd[k])) * W[k,n] + bias[n] )
// A is K-contiguous (row stride asm0). W has arbitrary strides (wsk, wsn).
// 256 threads = 8 waves; block computes a 64x64 tile; each wave computes one
// 16-row subtile x two 16-col subtiles via v_wmma_f32_16x16x32_bf16.
// f32 -> bf16 conversion happens once per element during LDS staging.
// LDS layouts: As (m,k) row-major 64x32 bf16; Bs (n,k) row-major 64x32 bf16.
// Per CDNA5 ISA 7.12.2 (wave32) the A fragment halves (K=8h..8h+7, +16) and
// the B fragment halves (K=8h..8h+7, +16 of column n) are contiguous -> ds_load_b128.
// Requires: M%64==0, N%64==0, K%32==0 (true for every call in this model).
// ---------------------------------------------------------------------------
__global__ __launch_bounds__(256) void wmma_gemm_kernel(
    const float* __restrict__ A, long asm0,
    const float* __restrict__ W, long wsk, long wsn,
    float* __restrict__ Out, long osm, long osn,
    const float* __restrict__ bias,
    const float* __restrict__ aadd,
    int M, int N, int K, int act)
{
    __shared__ __bf16 As[TM * TK];   // 4 KB
    __shared__ __bf16 Bs[TN * TK];   // 4 KB

    const int tid  = threadIdx.x;
    const int lane = tid & 31;
    const int wave = tid >> 5;
    const int nMac = N / TN;
    const int bm = blockIdx.x / nMac;
    const int bn = blockIdx.x - bm * nMac;
    const int m_base = bm * TM;
    const int n_base = bn * TN;

    // staging coordinates: 256 threads cover 64 rows x 4 k-octets (8 k each)
    const int sr = tid & 63;                 // tile row (m for A, n for B)
    const int sq = tid >> 6;                 // k-octet: k = 8*sq .. 8*sq+7

    // compute coordinates
    const int tmsub  = wave & 3;             // m subtile 0..3
    const int tnsub0 = (wave >> 2) * 2;      // n subtiles {tnsub0, tnsub0+1}
    const int l16 = lane & 15, hf = lane >> 4;

    v8f acc0 = {0.f, 0.f, 0.f, 0.f, 0.f, 0.f, 0.f, 0.f};
    v8f acc1 = acc0;

    const float* aptr = A + (long)(m_base + sr) * asm0 + 8 * sq; // K-contiguous
    const float* wptr = W + (long)(n_base + sr) * wsn;

    for (int k0 = 0; k0 < K; k0 += TK) {
        // ---------------- stage A tile (64 x 32) ----------------
        {
            const float* p = aptr + k0;
            float f[8];
#pragma unroll
            for (int j = 0; j < 8; ++j) f[j] = p[j];
            if (aadd) {
#pragma unroll
                for (int j = 0; j < 8; ++j) f[j] += aadd[k0 + 8 * sq + j];
            }
            v8bf h;
#pragma unroll
            for (int j = 0; j < 8; ++j) h[j] = (__bf16)f[j];
            *(v8bf*)&As[sr * TK + 8 * sq] = h;                  // ds_store_b128
        }
        // ---------------- stage B tile (n-major 64 x 32) ----------------
        {
            float f[8];
            if (wsk == 1) {                                      // K-contiguous W
                const float* p = wptr + (k0 + 8 * sq);
#pragma unroll
                for (int j = 0; j < 8; ++j) f[j] = p[j];
            } else {                                             // N-contiguous W
                const float* p = wptr + (long)(k0 + 8 * sq) * wsk;
#pragma unroll
                for (int j = 0; j < 8; ++j) f[j] = p[(long)j * wsk];
            }
            v8bf h;
#pragma unroll
            for (int j = 0; j < 8; ++j) h[j] = (__bf16)f[j];
            *(v8bf*)&Bs[sr * TK + 8 * sq] = h;                  // ds_store_b128
        }
        if (k0 + TK < K) {
            __builtin_prefetch(aptr + k0 + TK, 0, 3);           // global_prefetch_b8
            __builtin_prefetch(wptr + (long)(k0 + TK) * wsk, 0, 3);
        }
        __syncthreads();

        // ---------------- fragments from LDS (ds_load_b128) ----------------
        v16bf av, b0, b1;
        {
            const int ar = (tmsub * 16 + l16) * TK + 8 * hf;
            v8bf lo = *(const v8bf*)&As[ar];
            v8bf hi = *(const v8bf*)&As[ar + 16];
#pragma unroll
            for (int j = 0; j < 8; ++j) { av[j] = lo[j]; av[8 + j] = hi[j]; }
        }
        {
            const int br0 = (tnsub0 * 16 + l16) * TK + 8 * hf;
            v8bf lo = *(const v8bf*)&Bs[br0];
            v8bf hi = *(const v8bf*)&Bs[br0 + 16];
#pragma unroll
            for (int j = 0; j < 8; ++j) { b0[j] = lo[j]; b0[8 + j] = hi[j]; }
            const int br1 = ((tnsub0 + 1) * 16 + l16) * TK + 8 * hf;
            v8bf lo1 = *(const v8bf*)&Bs[br1];
            v8bf hi1 = *(const v8bf*)&Bs[br1 + 16];
#pragma unroll
            for (int j = 0; j < 8; ++j) { b1[j] = lo1[j]; b1[8 + j] = hi1[j]; }
        }
        acc0 = __builtin_amdgcn_wmma_f32_16x16x32_bf16(false, av, false, b0,
                                                       (short)0, acc0, false, false);
        acc1 = __builtin_amdgcn_wmma_f32_16x16x32_bf16(false, av, false, b1,
                                                       (short)0, acc1, false, false);
        __syncthreads();
    }

    // ---------------- epilogue: bias + activation + strided store ----------------
    const int n0g = n_base + tnsub0 * 16 + l16;
    const int n1g = n0g + 16;
    const float bv0 = bias ? bias[n0g] : 0.0f;
    const float bv1 = bias ? bias[n1g] : 0.0f;
#pragma unroll
    for (int r = 0; r < 8; ++r) {
        const int m = m_base + tmsub * 16 + 8 * hf + r;
        float v0 = acc0[r] + bv0;
        float v1 = acc1[r] + bv1;
        if (act == 1) {
            v0 = v0 * (1.0f / (1.0f + __expf(-v0)));
            v1 = v1 * (1.0f / (1.0f + __expf(-v1)));
        }
        Out[(long)m * osm + (long)n0g * osn] = v0;
        Out[(long)m * osm + (long)n1g * osn] = v1;
    }
}

// ---------------------------------------------------------------------------
// LayerNorm over last dim (D), one block (256 thr) per row. eps = 1e-5, biased var.
// ---------------------------------------------------------------------------
__global__ void ln_kernel(const float* __restrict__ x, const float* __restrict__ g,
                          const float* __restrict__ b, float* __restrict__ y, int Dn)
{
    long row = blockIdx.x;
    const float* xr = x + row * Dn;
    float* yr = y + row * Dn;
    __shared__ float s1[256], s2[256];
    int tid = threadIdx.x;
    float s = 0.f, q = 0.f;
    for (int c = tid; c < Dn; c += blockDim.x) { float v = xr[c]; s += v; q += v * v; }
    s1[tid] = s; s2[tid] = q; __syncthreads();
    for (int st = 128; st > 0; st >>= 1) {
        if (tid < st) { s1[tid] += s1[tid + st]; s2[tid] += s2[tid + st]; }
        __syncthreads();
    }
    float mean = s1[0] / (float)Dn;
    float var  = s2[0] / (float)Dn - mean * mean;
    float inv  = rsqrtf(var + 1e-5f);
    for (int c = tid; c < Dn; c += blockDim.x)
        yr[c] = (xr[c] - mean) * inv * g[c] + b[c];
}

// x += alpha * t
__global__ void axpy_kernel(float* __restrict__ x, const float* __restrict__ t,
                            float alpha, long n)
{
    long i = (long)blockIdx.x * blockDim.x + threadIdx.x;
    if (i < n) x[i] += alpha * t[i];
}

// sinusoidal positional encoding, one block per position
__global__ void pe_kernel(float* __restrict__ pe, int S, int Dn)
{
    int s = blockIdx.x;
    const float neglog = -logf(10000.0f) / (float)Dn;
    for (int i = threadIdx.x; i < (Dn >> 1); i += blockDim.x) {
        float div = __expf((float)(2 * i) * neglog);
        float ang = (float)s * div;
        pe[(long)s * Dn + 2 * i]     = sinf(ang);
        pe[(long)s * Dn + 2 * i + 1] = cosf(ang);
    }
}

// ---------------------------------------------------------------------------
// Fused Transformer-XL rel-shift + add + scale + softmax, in place on `score`.
// shifted[i,j] = psraw[i, S-1-i+j] (j<=i) ; 0 (j==i+1) ; psraw[i+1, j-i-2] (j>i+1)
// ---------------------------------------------------------------------------
__global__ void softmax_shift_kernel(float* __restrict__ score,
                                     const float* __restrict__ psraw,
                                     int S, float scale)
{
    int i   = blockIdx.x;
    int tid = threadIdx.x;
    float* row = score + (long)i * S;
    const float* p0 = psraw + (long)i * S;
    const float* p1 = p0 + S;                  // only dereferenced when i+1 < S
    __shared__ float red[256];
    float vloc[8];
    int cnt = 0;
    float vmax = -3.4e38f;
    for (int j = tid; j < S; j += blockDim.x) {
        float p;
        if (j <= i)            p = p0[S - 1 - i + j];
        else if (j == i + 1)   p = 0.0f;
        else                   p = p1[j - i - 2];
        float v = (row[j] + p) * scale;
        vloc[cnt++] = v;
        vmax = fmaxf(vmax, v);
    }
    red[tid] = vmax; __syncthreads();
    for (int st = 128; st > 0; st >>= 1) {
        if (tid < st) red[tid] = fmaxf(red[tid], red[tid + st]);
        __syncthreads();
    }
    vmax = red[0]; __syncthreads();
    float sum = 0.f; cnt = 0;
    for (int j = tid; j < S; j += blockDim.x) {
        float e = __expf(vloc[cnt] - vmax);
        vloc[cnt++] = e; sum += e;
    }
    red[tid] = sum; __syncthreads();
    for (int st = 128; st > 0; st >>= 1) {
        if (tid < st) red[tid] += red[tid + st];
        __syncthreads();
    }
    float inv = 1.0f / red[0];
    cnt = 0;
    for (int j = tid; j < S; j += blockDim.x) row[j] = vloc[cnt++] * inv;
}

// GLU: out = a * sigmoid(gate)  (split of (B,S,2*INNER) along last dim)
__global__ void glu_kernel(const float* __restrict__ c, float* __restrict__ o, long total)
{
    long t = (long)blockIdx.x * blockDim.x + threadIdx.x;
    if (t >= total) return;
    long bs = t / INNERD;
    int  ch = (int)(t - bs * INNERD);
    float a  = c[bs * (2 * INNERD) + ch];
    float gt = c[bs * (2 * INNERD) + INNERD + ch];
    o[t] = a * (1.0f / (1.0f + __expf(-gt)));
}

// depthwise conv (k=31, same pad) + BN + swish, layout (B,S,INNER)
__global__ void dwconv_kernel(const float* __restrict__ in, const float* __restrict__ w,
                              const float* __restrict__ db, const float* __restrict__ g,
                              const float* __restrict__ bb2, const float* __restrict__ mean,
                              const float* __restrict__ var, float* __restrict__ o, long total)
{
    long t = (long)blockIdx.x * blockDim.x + threadIdx.x;
    if (t >= total) return;
    int  ch = (int)(t % INNERD);
    long bs = t / INNERD;
    int  s  = (int)(bs % SS);
    long b  = bs / SS;
    const float* base = in + (b * SS) * (long)INNERD + ch;
    float acc = 0.f;
#pragma unroll
    for (int k = 0; k < KW; ++k) {
        int sp = s + k - (KW / 2);
        if (sp >= 0 && sp < SS) acc += base[(long)sp * INNERD] * w[ch * KW + k];
    }
    acc += db[ch];
    acc = (acc - mean[ch]) * (g[ch] * rsqrtf(var[ch] + 1e-5f)) + bb2[ch];
    o[t] = acc * (1.0f / (1.0f + __expf(-acc)));
}

// ---------------------------------------------------------------------------
static void gemm(const float* A, long asm0,
                 const float* W, long wsk, long wsn,
                 float* Out, long osm, long osn,
                 const float* bias, const float* aadd,
                 int M, int N, int K, int act, hipStream_t s)
{
    int grid = (M / TM) * (N / TN);
    wmma_gemm_kernel<<<grid, 256, 0, s>>>(A, asm0, W, wsk, wsn,
                                          Out, osm, osn, bias, aadd, M, N, K, act);
}

extern "C" void kernel_launch(void* const* d_in, const int* in_sizes, int n_in,
                              void* d_out, int out_size, void* d_ws, size_t ws_size,
                              hipStream_t stream)
{
    (void)in_sizes; (void)n_in; (void)out_size; (void)ws_size;
    const float* x_in      = (const float*)d_in[0];
    const float* ff1_ln_g  = (const float*)d_in[1];
    const float* ff1_ln_b  = (const float*)d_in[2];
    const float* ff1_w1    = (const float*)d_in[3];
    const float* ff1_b1    = (const float*)d_in[4];
    const float* ff1_w2    = (const float*)d_in[5];
    const float* ff1_b2    = (const float*)d_in[6];
    const float* attn_ln_g = (const float*)d_in[7];
    const float* attn_ln_b = (const float*)d_in[8];
    const float* wq        = (const float*)d_in[9];
    const float* bq        = (const float*)d_in[10];
    const float* wk        = (const float*)d_in[11];
    const float* bk        = (const float*)d_in[12];
    const float* wv        = (const float*)d_in[13];
    const float* bv_p      = (const float*)d_in[14];
    const float* wpos      = (const float*)d_in[15];
    const float* u_bias    = (const float*)d_in[16];
    const float* v_bias    = (const float*)d_in[17];
    const float* wo        = (const float*)d_in[18];
    const float* bo        = (const float*)d_in[19];
    const float* conv_ln_g = (const float*)d_in[20];
    const float* conv_ln_b = (const float*)d_in[21];
    const float* pw1_w     = (const float*)d_in[22];
    const float* pw1_b     = (const float*)d_in[23];
    const float* dw_w      = (const float*)d_in[24];
    const float* dw_b      = (const float*)d_in[25];
    const float* bn_g      = (const float*)d_in[26];
    const float* bn_b      = (const float*)d_in[27];
    const float* bn_mean   = (const float*)d_in[28];
    const float* bn_var    = (const float*)d_in[29];
    const float* pw2_w     = (const float*)d_in[30];
    const float* pw2_b     = (const float*)d_in[31];
    const float* ff2_ln_g  = (const float*)d_in[32];
    const float* ff2_ln_b  = (const float*)d_in[33];
    const float* ff2_w1    = (const float*)d_in[34];
    const float* ff2_b1    = (const float*)d_in[35];
    const float* ff2_w2    = (const float*)d_in[36];
    const float* ff2_b2    = (const float*)d_in[37];
    const float* norm_g    = (const float*)d_in[38];
    const float* norm_b    = (const float*)d_in[39];
    float* out = (float*)d_out;

    // workspace carve-up (floats)
    float* ws = (float*)d_ws;
    size_t off = 0;
    const size_t NTOK = (size_t)BB * SS;
    float* xbuf   = ws + off; off += NTOK * DD;       // running residual x
    float* ybuf   = ws + off; off += NTOK * DD;       // LN output
    float* tbuf   = ws + off; off += NTOK * DD;       // pre-residual GEMM out
    float* ffbuf  = ws + off; off += NTOK * FFD;      // FF hidden / conv pw1 out
    float* qbuf   = ws + off; off += NTOK * DD;
    float* kbuf   = ws + off; off += NTOK * DD;
    float* vbuf   = ws + off; off += NTOK * DD;
    float* ctxbuf = ws + off; off += NTOK * DD;
    float* pebuf  = ws + off; off += (size_t)SS * DD;
    float* posbuf = ws + off; off += (size_t)SS * DD;
    float* score  = ws + off; off += (size_t)SS * SS; // per-(b,h), reused
    float* psraw  = ws + off; off += (size_t)SS * SS; // per-(b,h), reused
    float* glubuf = ws + off; off += NTOK * INNERD;
    float* convbf = ws + off; off += NTOK * INNERD;

    const int  ROWS = (int)NTOK;
    const long NTOT = (long)NTOK * DD;
    const int  gETok = (int)((NTOT + 255) / 256);
    const long NINN = (long)NTOK * INNERD;
    const int  gEInn = (int)((NINN + 255) / 256);

    hipMemcpyAsync(xbuf, x_in, NTOT * sizeof(float), hipMemcpyDeviceToDevice, stream);

    // ---------------- FF1 (half residual) ----------------
    ln_kernel<<<ROWS, 256, 0, stream>>>(xbuf, ff1_ln_g, ff1_ln_b, ybuf, DD);
    gemm(ybuf, DD, ff1_w1, FFD, 1, ffbuf, FFD, 1, ff1_b1, nullptr, ROWS, FFD, DD, 1, stream);
    gemm(ffbuf, FFD, ff1_w2, DD, 1, tbuf, DD, 1, ff1_b2, nullptr, ROWS, DD, FFD, 0, stream);
    axpy_kernel<<<gETok, 256, 0, stream>>>(xbuf, tbuf, 0.5f, NTOT);

    // ---------------- relative-position MHSA ----------------
    ln_kernel<<<ROWS, 256, 0, stream>>>(xbuf, attn_ln_g, attn_ln_b, ybuf, DD);
    gemm(ybuf, DD, wq, DD, 1, qbuf, DD, 1, bq,   nullptr, ROWS, DD, DD, 0, stream);
    gemm(ybuf, DD, wk, DD, 1, kbuf, DD, 1, bk,   nullptr, ROWS, DD, DD, 0, stream);
    gemm(ybuf, DD, wv, DD, 1, vbuf, DD, 1, bv_p, nullptr, ROWS, DD, DD, 0, stream);
    pe_kernel<<<SS, 256, 0, stream>>>(pebuf, SS, DD);
    gemm(pebuf, DD, wpos, DD, 1, posbuf, DD, 1, nullptr, nullptr, SS, DD, DD, 0, stream);

    const float scale = 1.0f / sqrtf((float)DD);
    for (int b = 0; b < BB; ++b) {
        for (int h = 0; h < HH; ++h) {
            const float* qh = qbuf + (size_t)b * SS * DD + (size_t)h * DHH;
            const float* kh = kbuf + (size_t)b * SS * DD + (size_t)h * DHH;
            const float* vh = vbuf + (size_t)b * SS * DD + (size_t)h * DHH;
            float*       ch = ctxbuf + (size_t)b * SS * DD + (size_t)h * DHH;
            // score[s,t] = (q+u_bias) . k   (W = K^T via strides: wsk=1, wsn=D)
            gemm(qh, DD, kh, 1, DD, score, SS, 1, nullptr, u_bias + h * DHH,
                 SS, SS, DHH, 0, stream);
            // psraw[s,t] = (q+v_bias) . pos
            gemm(qh, DD, posbuf + h * DHH, 1, DD, psraw, SS, 1, nullptr, v_bias + h * DHH,
                 SS, SS, DHH, 0, stream);
            softmax_shift_kernel<<<SS, 256, 0, stream>>>(score, psraw, SS, scale);
            // ctx[s,d] = attn @ v   (wsk=D, wsn=1)
            gemm(score, SS, vh, DD, 1, ch, DD, 1, nullptr, nullptr,
                 SS, DHH, SS, 0, stream);
        }
    }
    gemm(ctxbuf, DD, wo, DD, 1, tbuf, DD, 1, bo, nullptr, ROWS, DD, DD, 0, stream);
    axpy_kernel<<<gETok, 256, 0, stream>>>(xbuf, tbuf, 1.0f, NTOT);

    // ---------------- conformer conv module ----------------
    ln_kernel<<<ROWS, 256, 0, stream>>>(xbuf, conv_ln_g, conv_ln_b, ybuf, DD);
    gemm(ybuf, DD, pw1_w, 2 * INNERD, 1, ffbuf, 2 * INNERD, 1, pw1_b, nullptr,
         ROWS, 2 * INNERD, DD, 0, stream);
    glu_kernel<<<gEInn, 256, 0, stream>>>(ffbuf, glubuf, NINN);
    dwconv_kernel<<<gEInn, 256, 0, stream>>>(glubuf, dw_w, dw_b, bn_g, bn_b,
                                             bn_mean, bn_var, convbf, NINN);
    gemm(convbf, INNERD, pw2_w, DD, 1, tbuf, DD, 1, pw2_b, nullptr,
         ROWS, DD, INNERD, 0, stream);
    axpy_kernel<<<gETok, 256, 0, stream>>>(xbuf, tbuf, 1.0f, NTOT);

    // ---------------- FF2 (half residual) ----------------
    ln_kernel<<<ROWS, 256, 0, stream>>>(xbuf, ff2_ln_g, ff2_ln_b, ybuf, DD);
    gemm(ybuf, DD, ff2_w1, FFD, 1, ffbuf, FFD, 1, ff2_b1, nullptr, ROWS, FFD, DD, 1, stream);
    gemm(ffbuf, FFD, ff2_w2, DD, 1, tbuf, DD, 1, ff2_b2, nullptr, ROWS, DD, FFD, 0, stream);
    axpy_kernel<<<gETok, 256, 0, stream>>>(xbuf, tbuf, 0.5f, NTOT);

    // ---------------- final LN -> output ----------------
    ln_kernel<<<ROWS, 256, 0, stream>>>(xbuf, norm_g, norm_b, out, DD);
}
